// TiSASRec_87875030876321
// MI455X (gfx1250) — compile-verified, matrix operand order checked by reading, not run on hardware
//
#include <hip/hip_runtime.h>
#include <hip/hip_bf16.h>
#include <math.h>

// ---------------- problem constants ----------------
#define B_     2
#define S_     512
#define H_     128
#define NH_    4
#define D_     32
#define NB_    2
#define BS_    (B_*S_)
#define NMETA_ 16
#define BAG_   5
#define TS_    257        // TSPAN+1
#define TSP_   288        // TS_ padded to multiple of 32 (WMMA K chunks)
#define QTLD_  264        // per-head qtK row stride (257 padded)
#define NEGV_  (-4294967295.0f)
#define EPS_EMB_ 1e-5f
#define EPS_     1e-8f

typedef __attribute__((ext_vector_type(16))) _Float16 v16h;
typedef __attribute__((ext_vector_type(8)))  _Float16 v8h;
typedef __attribute__((ext_vector_type(8)))  float    v8f;

// Per-lane WMMA 16-bit fragment load matching the documented CDNA5 layout:
// lane L: row = L&15, kb = (L>>4)*8 ; holds K = {kb..kb+7, kb+16..kb+23}.
// Two contiguous 16-byte loads.
__device__ __forceinline__ v16h frag_ld(const _Float16* p) {
  v8h lo = *(const v8h*)p;
  v8h hi = *(const v8h*)(p + 16);
  v16h r;
#pragma unroll
  for (int i = 0; i < 8; ++i) { r[i] = lo[i]; r[8 + i] = hi[i]; }
  return r;
}

// ---------------- small utility kernels ----------------
__global__ void f16cvt_kernel(const float* __restrict__ in, _Float16* __restrict__ out, int n) {
  int i = blockIdx.x * blockDim.x + threadIdx.x;
  if (i < n) out[i] = (_Float16)in[i];
}

__global__ void add_kernel(float* __restrict__ dst, const float* __restrict__ a,
                           const float* __restrict__ b, int n) {
  int i = blockIdx.x * blockDim.x + threadIdx.x;
  if (i < n) dst[i] = a[i] + b[i];
}

// v16 [B*S][H] -> vt16 [B][NH][D][S]  (so AV B-fragments load contiguously along keys)
__global__ void transpose_v_kernel(const _Float16* __restrict__ v, _Float16* __restrict__ vt) {
  int idx = blockIdx.x * blockDim.x + threadIdx.x;
  if (idx >= B_ * NH_ * D_ * S_) return;
  int j = idx % S_; int t = idx / S_;
  int d = t % D_;   t /= D_;
  int h = t % NH_;  int b = t / NH_;
  vt[idx] = v[((size_t)(b * S_ + j)) * H_ + h * D_ + d];
}

// pos_V_w [S][H] (f32) -> pvt16 [NH][D][S]
__global__ void transpose_posv_kernel(const float* __restrict__ pv, _Float16* __restrict__ pvt) {
  int idx = blockIdx.x * blockDim.x + threadIdx.x;
  if (idx >= NH_ * D_ * S_) return;
  int j = idx % S_; int t = idx / S_;
  int d = t % D_;   int h = t / D_;
  pvt[idx] = (_Float16)pv[(size_t)j * H_ + h * D_ + d];
}

// time_V_w [257][H] (f32) -> tvt16 [NH][D][288] (zero padded)
__global__ void transpose_tv_kernel(const float* __restrict__ tv, _Float16* __restrict__ tvt) {
  int idx = blockIdx.x * blockDim.x + threadIdx.x;
  if (idx >= NH_ * D_ * TSP_) return;
  int tt = idx % TSP_; int t = idx / TSP_;
  int d = t % D_;      int h = t / D_;
  tvt[idx] = (tt < TS_) ? (_Float16)tv[(size_t)tt * H_ + h * D_ + d] : (_Float16)0.0f;
}

// ---------------- seq2vec: build comb[B*S][2H] in f16 ----------------
__global__ void build_comb_kernel(const int* __restrict__ ids,
                                  const float* __restrict__ meta,
                                  const int* __restrict__ cats,
                                  const float* __restrict__ item_w,
                                  const float* __restrict__ cat_w,
                                  const float* __restrict__ numW,
                                  const float* __restrict__ numB,
                                  _Float16* __restrict__ comb16) {
  int t = blockIdx.x;          // token 0..BS-1
  int c = threadIdx.x;         // 0..127
  int id = ids[t];
  const float sqrtH = 11.313708498984761f;   // sqrt(128)
  comb16[(size_t)t * 256 + c] = (_Float16)(item_w[(size_t)id * H_ + c] * sqrtH);
  if (c < 64) {
    // numeric projection: meta[t,:16] @ num_W[c,:16]^T + b
    float acc = numB[c];
#pragma unroll
    for (int m = 0; m < NMETA_; ++m) acc += meta[(size_t)t * NMETA_ + m] * numW[(size_t)c * NMETA_ + m];
    comb16[(size_t)t * 256 + 128 + c] = (_Float16)acc;
    // embedding-bag mean (padding_idx=0)
    float cs = 0.0f; int cnt = 0;
#pragma unroll
    for (int g = 0; g < BAG_; ++g) {
      int ci = cats[(size_t)t * BAG_ + g];
      if (ci != 0) { cs += cat_w[(size_t)ci * 64 + c]; cnt++; }
    }
    comb16[(size_t)t * 256 + 192 + c] = (_Float16)(cs / (float)max(cnt, 1));
  }
}

// ---------------- generic WMMA GEMM: C[M,N] = A[M,K] @ W[N,K]^T ----------------
// One wave per 16x16 C tile; f16 inputs, f32 accumulate; optional epilogues.
enum { GEMM_GELU = 1, GEMM_RESID = 2, GEMM_MASK = 4 };

__global__ void __launch_bounds__(32)
gemm_wmma_kernel(const _Float16* __restrict__ A, int lda,
                 const _Float16* __restrict__ W, int ldb,
                 const float* __restrict__ bias,
                 float* __restrict__ C32, _Float16* __restrict__ C16, int ldc,
                 int M, int N, int K, int flags,
                 const float* __restrict__ resid, const int* __restrict__ ids) {
  int lane = threadIdx.x & 31;
  int mt = blockIdx.x, nt = blockIdx.y;
  int r16 = lane & 15;
  int kb  = (lane >> 4) * 8;
  int am = mt * 16 + r16; if (am > M - 1) am = M - 1;   // clamp (N=257 tail tiles)
  int bn = nt * 16 + r16; if (bn > N - 1) bn = N - 1;
  const _Float16* Ap = A + (size_t)am * lda + kb;
  const _Float16* Wp = W + (size_t)bn * ldb + kb;
  v8f acc = {};
  for (int k = 0; k < K; k += 32) {
    v16h a = frag_ld(Ap + k);
    v16h b = frag_ld(Wp + k);
    acc = __builtin_amdgcn_wmma_f32_16x16x32_f16(false, a, false, b, (short)0, acc, false, false);
  }
  // C layout: VGPR r -> lanes 0-15: M=r ; lanes 16-31: M=8+r ; N = lane&15
  int hi8 = (lane >> 4) * 8;
  int col = nt * 16 + (lane & 15);
#pragma unroll
  for (int r = 0; r < 8; ++r) {
    int row = mt * 16 + hi8 + r;
    if (row < M && col < N) {
      float v = acc[r];
      if (bias) v += bias[col];
      if (flags & GEMM_GELU) v = 0.5f * v * (1.0f + erff(v * 0.70710678118654752f));
      if (flags & GEMM_RESID) v += resid[(size_t)row * ldc + col];
      if (flags & GEMM_MASK)  { if (ids[row] == 0) v = 0.0f; }
      if (C32) C32[(size_t)row * ldc + col] = v;
      if (C16) C16[(size_t)row * ldc + col] = (_Float16)v;
    }
  }
}

// ---------------- LayerNorm over H=128 (one block of 128 per token) ----------------
__global__ void ln_kernel(const float* __restrict__ x, const float* __restrict__ g,
                          const float* __restrict__ be, float eps,
                          const int* __restrict__ ids,
                          float* __restrict__ y32, _Float16* __restrict__ y16) {
  __shared__ float red[128];
  int t = blockIdx.x, c = threadIdx.x;
  float v = x[(size_t)t * H_ + c];
  red[c] = v; __syncthreads();
  for (int s = 64; s > 0; s >>= 1) { if (c < s) red[c] += red[c + s]; __syncthreads(); }
  float mean = red[0] * (1.0f / H_); __syncthreads();
  float d = v - mean;
  red[c] = d * d; __syncthreads();
  for (int s = 64; s > 0; s >>= 1) { if (c < s) red[c] += red[c + s]; __syncthreads(); }
  float var = red[0] * (1.0f / H_);
  float o = d * rsqrtf(var + eps) * g[c] + be[c];
  if (ids && ids[t] == 0) o = 0.0f;
  if (y32) y32[(size_t)t * H_ + c] = o;
  if (y16) y16[(size_t)t * H_ + c] = (_Float16)o;
}

// ---------------- time-relative attention, one wave per (b,h,16-query tile) ----------------
__global__ void __launch_bounds__(32)
attn_wmma_kernel(const _Float16* __restrict__ q16, const _Float16* __restrict__ k16,
                 const _Float16* __restrict__ pk16, const _Float16* __restrict__ vt16,
                 const _Float16* __restrict__ pvt16, const _Float16* __restrict__ tvt16,
                 const float* __restrict__ qtk, const int* __restrict__ tmat,
                 const int* __restrict__ ids, float* __restrict__ out) {
  // Scores (f32) and later the time-bucket table share storage (48 KB total LDS).
  __shared__ union { float s[16][S_]; float w[16][TSP_]; } sh;
  __shared__ _Float16 sPh[16][S_];

  int lane = threadIdx.x & 31;
  int it = blockIdx.x, h = blockIdx.y, b = blockIdx.z;
  int r16 = lane & 15;
  int kb  = (lane >> 4) * 8;     // also the 8-row epilogue offset
  const float inv_sqrt_d = 0.17677669529663687f;  // 1/sqrt(32)

  // A-fragment: 16 query rows x D=32 (one WMMA covers the full head dim)
  v16h aQ = frag_ld(q16 + ((size_t)(b * S_ + it * 16 + r16)) * H_ + h * D_ + kb);

  // -------- pass 1: scores = (Q.K + Q.posK + qtK[gather]) * inv_sqrt_d, masked --------
  for (int jt = 0; jt < S_ / 16; ++jt) {
    if (jt + 1 < S_ / 16)
      __builtin_prefetch(k16 + ((size_t)(b * S_ + (jt + 1) * 16 + r16)) * H_ + h * D_ + kb, 0, 1);
    v16h bK = frag_ld(k16 + ((size_t)(b * S_ + jt * 16 + r16)) * H_ + h * D_ + kb);
    v16h bP = frag_ld(pk16 + ((size_t)(jt * 16 + r16)) * H_ + h * D_ + kb);
    v8f c = {};
    c = __builtin_amdgcn_wmma_f32_16x16x32_f16(false, aQ, false, bK, (short)0, c, false, false);
    c = __builtin_amdgcn_wmma_f32_16x16x32_f16(false, aQ, false, bP, (short)0, c, false, false);
#pragma unroll
    for (int r = 0; r < 8; ++r) {
      int il = kb + r;                         // local query row 0..15
      int i  = it * 16 + il;
      int j  = jt * 16 + (lane & 15);
      int tm = tmat[((size_t)(b * S_ + i)) * S_ + j];
      float qt = qtk[(((size_t)(b * S_ + i)) * NH_ + h) * QTLD_ + tm];
      float val = (c[r] + qt) * inv_sqrt_d;
      if (ids[b * S_ + i] == 0) val = NEGV_;   // timeline mask (query rows)
      if (j > i) val = NEGV_;                  // causal
      sh.s[il][j] = val;
    }
  }
  __syncthreads();

  // -------- pass 2: row softmax (wave32 shuffles), write P as f16 --------
  for (int rr = 0; rr < 16; ++rr) {
    float mx = -3.4e38f;
    for (int c0 = lane; c0 < S_; c0 += 32) mx = fmaxf(mx, sh.s[rr][c0]);
#pragma unroll
    for (int o = 16; o > 0; o >>= 1) mx = fmaxf(mx, __shfl_xor(mx, o, 32));
    float sum = 0.0f;
    for (int c0 = lane; c0 < S_; c0 += 32) {
      float e = __expf(sh.s[rr][c0] - mx);
      sh.s[rr][c0] = e; sum += e;
    }
#pragma unroll
    for (int o = 16; o > 0; o >>= 1) sum += __shfl_xor(sum, o, 32);
    float inv = 1.0f / sum;
    for (int c0 = lane; c0 < S_; c0 += 32)
      sPh[rr][c0] = (_Float16)(sh.s[rr][c0] * inv);
  }
  __syncthreads();

  // -------- pass 3: time-bucket histogram W[row][t] += p  (ds_add_f32 atomics) --------
  for (int i2 = lane; i2 < 16 * TSP_; i2 += 32) ((float*)sh.w)[i2] = 0.0f;
  __syncthreads();
  for (int rr = 0; rr < 16; ++rr) {
    int i = it * 16 + rr;
    for (int c0 = lane; c0 < S_; c0 += 32) {
      int tm = tmat[((size_t)(b * S_ + i)) * S_ + c0];
      atomicAdd(&sh.w[rr][tm], (float)sPh[rr][c0]);
    }
  }
  __syncthreads();
  // in-place f32 -> f16 (safe: per-wave loads complete before stores each step)
  {
    _Float16* wh = (_Float16*)&sh.w[0][0];
    const float* wf = (const float*)&sh.w[0][0];
    for (int i2 = lane; i2 < 16 * TSP_; i2 += 32) wh[i2] = (_Float16)wf[i2];
  }
  __syncthreads();
  const _Float16* wh = (const _Float16*)&sh.w[0][0];

  // -------- pass 4: out = P@(V) + P@(posV) + W@(time_V) via chained WMMAs --------
#pragma unroll
  for (int ns = 0; ns < 2; ++ns) {
    v8f acc = {};
    int d = ns * 16 + r16;                       // B-fragment row index (head dim)
    for (int kt = 0; kt < S_ / 32; ++kt) {
      v16h aP = frag_ld(&sPh[r16][kt * 32 + kb]);
      v16h bV = frag_ld(vt16 + ((size_t)((b * NH_ + h) * D_ + d)) * S_ + kt * 32 + kb);
      v16h bQ = frag_ld(pvt16 + ((size_t)(h * D_ + d)) * S_ + kt * 32 + kb);
      acc = __builtin_amdgcn_wmma_f32_16x16x32_f16(false, aP, false, bV, (short)0, acc, false, false);
      acc = __builtin_amdgcn_wmma_f32_16x16x32_f16(false, aP, false, bQ, (short)0, acc, false, false);
    }
    for (int kt = 0; kt < TSP_ / 32; ++kt) {
      v16h aW = frag_ld(wh + (size_t)r16 * TSP_ + kt * 32 + kb);
      v16h bT = frag_ld(tvt16 + ((size_t)(h * D_ + d)) * TSP_ + kt * 32 + kb);
      acc = __builtin_amdgcn_wmma_f32_16x16x32_f16(false, aW, false, bT, (short)0, acc, false, false);
    }
#pragma unroll
    for (int r = 0; r < 8; ++r) {
      int i  = it * 16 + kb + r;
      int dd = ns * 16 + (lane & 15);
      out[((size_t)(b * S_ + i)) * H_ + h * D_ + dd] = acc[r];
    }
  }
}

// ---------------- host-side orchestration ----------------
extern "C" void kernel_launch(void* const* d_in, const int* in_sizes, int n_in,
                              void* d_out, int out_size, void* d_ws, size_t ws_size,
                              hipStream_t stream) {
  (void)in_sizes; (void)n_in; (void)out_size; (void)ws_size;
  const int*   ids      = (const int*)  d_in[0];
  const float* meta     = (const float*)d_in[1];
  const int*   cats     = (const int*)  d_in[2];
  const int*   tmat     = (const int*)  d_in[3];
  const float* item_w   = (const float*)d_in[4];
  const float* cat_w    = (const float*)d_in[5];
  const float* numW     = (const float*)d_in[6];
  const float* numB     = (const float*)d_in[7];
  const float* fusionW  = (const float*)d_in[8];
  const float* fusionB  = (const float*)d_in[9];
  const float* embG     = (const float*)d_in[10];
  const float* embB     = (const float*)d_in[11];
  const float* posK     = (const float*)d_in[12];
  const float* posV     = (const float*)d_in[13];
  const float* timeK    = (const float*)d_in[14];
  const float* timeV    = (const float*)d_in[15];
  const float* attnG    = (const float*)d_in[16];
  const float* attnB    = (const float*)d_in[17];
  const float* QW       = (const float*)d_in[18];
  const float* Qb       = (const float*)d_in[19];
  const float* KW       = (const float*)d_in[20];
  const float* Kb       = (const float*)d_in[21];
  const float* VW       = (const float*)d_in[22];
  const float* Vb       = (const float*)d_in[23];
  const float* fwdG     = (const float*)d_in[24];
  const float* fwdB     = (const float*)d_in[25];
  const float* w1W      = (const float*)d_in[26];
  const float* w1b      = (const float*)d_in[27];
  const float* w2W      = (const float*)d_in[28];
  const float* w2b      = (const float*)d_in[29];
  const float* lastG    = (const float*)d_in[30];
  const float* lastB    = (const float*)d_in[31];

  // workspace carve-up (256B aligned)
  char* p = (char*)d_ws;
  auto alloc = [&](size_t bytes) -> void* {
    void* r = (void*)p; p += (bytes + 255) & ~(size_t)255; return r;
  };
  _Float16* fW16   = (_Float16*)alloc((size_t)H_ * 2 * H_ * 2);
  _Float16* qW16   = (_Float16*)alloc((size_t)NB_ * H_ * H_ * 2);
  _Float16* kW16   = (_Float16*)alloc((size_t)NB_ * H_ * H_ * 2);
  _Float16* vW16   = (_Float16*)alloc((size_t)NB_ * H_ * H_ * 2);
  _Float16* w1W16  = (_Float16*)alloc((size_t)NB_ * H_ * H_ * 2);
  _Float16* w2W16  = (_Float16*)alloc((size_t)NB_ * H_ * H_ * 2);
  _Float16* pK16   = (_Float16*)alloc((size_t)S_ * H_ * 2);
  _Float16* tK16   = (_Float16*)alloc((size_t)TS_ * H_ * 2);
  _Float16* pvt16  = (_Float16*)alloc((size_t)NH_ * D_ * S_ * 2);
  _Float16* tvt16  = (_Float16*)alloc((size_t)NH_ * D_ * TSP_ * 2);
  _Float16* comb16 = (_Float16*)alloc((size_t)BS_ * 2 * H_ * 2);
  float*    fused32= (float*)   alloc((size_t)BS_ * H_ * 4);
  float*    vecs32 = (float*)   alloc((size_t)BS_ * H_ * 4);
  _Float16* vecs16 = (_Float16*)alloc((size_t)BS_ * H_ * 2);
  float*    qn32   = (float*)   alloc((size_t)BS_ * H_ * 4);
  _Float16* qn16   = (_Float16*)alloc((size_t)BS_ * H_ * 2);
  _Float16* q16    = (_Float16*)alloc((size_t)BS_ * H_ * 2);
  _Float16* k16    = (_Float16*)alloc((size_t)BS_ * H_ * 2);
  _Float16* v16b   = (_Float16*)alloc((size_t)BS_ * H_ * 2);
  _Float16* vt16   = (_Float16*)alloc((size_t)BS_ * H_ * 2);
  float*    qtk    = (float*)   alloc((size_t)BS_ * NH_ * QTLD_ * 4);
  float*    aout32 = (float*)   alloc((size_t)BS_ * H_ * 4);
  float*    pre32  = (float*)   alloc((size_t)BS_ * H_ * 4);
  _Float16* h1_16  = (_Float16*)alloc((size_t)BS_ * H_ * 2);

  auto cvt = [&](const float* in, _Float16* out, int n) {
    f16cvt_kernel<<<(n + 255) / 256, 256, 0, stream>>>(in, out, n);
  };
  // weight down-conversion (once per launch; deterministic)
  cvt(fusionW, fW16, H_ * 2 * H_);
  cvt(QW,  qW16,  NB_ * H_ * H_);
  cvt(KW,  kW16,  NB_ * H_ * H_);
  cvt(VW,  vW16,  NB_ * H_ * H_);
  cvt(w1W, w1W16, NB_ * H_ * H_);
  cvt(w2W, w2W16, NB_ * H_ * H_);
  cvt(posK, pK16, S_ * H_);
  cvt(timeK, tK16, TS_ * H_);
  {
    int n = NH_ * D_ * S_;
    transpose_posv_kernel<<<(n + 255) / 256, 256, 0, stream>>>(posV, pvt16);
    n = NH_ * D_ * TSP_;
    transpose_tv_kernel<<<(n + 255) / 256, 256, 0, stream>>>(timeV, tvt16);
  }

  // ---- seq2vec ----
  build_comb_kernel<<<BS_, 128, 0, stream>>>(ids, meta, cats, item_w, cat_w, numW, numB, comb16);
  gemm_wmma_kernel<<<dim3(BS_ / 16, H_ / 16), 32, 0, stream>>>(
      comb16, 2 * H_, fW16, 2 * H_, fusionB, fused32, nullptr, H_,
      BS_, H_, 2 * H_, 0, nullptr, nullptr);
  ln_kernel<<<BS_, 128, 0, stream>>>(fused32, embG, embB, EPS_EMB_, ids, vecs32, vecs16);

  // ---- transformer blocks ----
  for (int blk = 0; blk < NB_; ++blk) {
    ln_kernel<<<BS_, 128, 0, stream>>>(vecs32, attnG + blk * H_, attnB + blk * H_, EPS_,
                                       nullptr, qn32, qn16);
    gemm_wmma_kernel<<<dim3(BS_ / 16, H_ / 16), 32, 0, stream>>>(
        qn16, H_, qW16 + (size_t)blk * H_ * H_, H_, Qb + blk * H_,
        nullptr, q16, H_, BS_, H_, H_, 0, nullptr, nullptr);
    gemm_wmma_kernel<<<dim3(BS_ / 16, H_ / 16), 32, 0, stream>>>(
        vecs16, H_, kW16 + (size_t)blk * H_ * H_, H_, Kb + blk * H_,
        nullptr, k16, H_, BS_, H_, H_, 0, nullptr, nullptr);
    gemm_wmma_kernel<<<dim3(BS_ / 16, H_ / 16), 32, 0, stream>>>(
        vecs16, H_, vW16 + (size_t)blk * H_ * H_, H_, Vb + blk * H_,
        nullptr, v16b, H_, BS_, H_, H_, 0, nullptr, nullptr);
    {
      int n = B_ * NH_ * D_ * S_;
      transpose_v_kernel<<<(n + 255) / 256, 256, 0, stream>>>(v16b, vt16);
    }
    // qtK[b,i,h,t] = Q_head . time_K_head^T   ([1024,32] x [257,32]^T per head)
    for (int h = 0; h < NH_; ++h) {
      gemm_wmma_kernel<<<dim3(BS_ / 16, (TS_ + 15) / 16), 32, 0, stream>>>(
          q16 + h * D_, H_, tK16 + h * D_, H_, nullptr,
          qtk + h * QTLD_, nullptr, NH_ * QTLD_,
          BS_, TS_, D_, 0, nullptr, nullptr);
    }
    attn_wmma_kernel<<<dim3(S_ / 16, NH_, B_), 32, 0, stream>>>(
        q16, k16, pK16, vt16, pvt16, tvt16, qtk, tmat, ids, aout32);
    add_kernel<<<(BS_ * H_ + 255) / 256, 256, 0, stream>>>(pre32, qn32, aout32, BS_ * H_);
    ln_kernel<<<BS_, 128, 0, stream>>>(pre32, fwdG + blk * H_, fwdB + blk * H_, EPS_,
                                       nullptr, vecs32, vecs16);
    gemm_wmma_kernel<<<dim3(BS_ / 16, H_ / 16), 32, 0, stream>>>(
        vecs16, H_, w1W16 + (size_t)blk * H_ * H_, H_, w1b + blk * H_,
        nullptr, h1_16, H_, BS_, H_, H_, GEMM_GELU, nullptr, nullptr);
    gemm_wmma_kernel<<<dim3(BS_ / 16, H_ / 16), 32, 0, stream>>>(
        h1_16, H_, w2W16 + (size_t)blk * H_ * H_, H_, w2b + blk * H_,
        vecs32, vecs16, H_, BS_, H_, H_, GEMM_RESID | GEMM_MASK, vecs32, ids);
  }

  // ---- final LayerNorm -> d_out (f32) ----
  ln_kernel<<<BS_, 128, 0, stream>>>(vecs32, lastG, lastB, EPS_, nullptr,
                                     (float*)d_out, nullptr);
}
